// CriticRNN_37039797961163
// MI455X (gfx1250) — matrix operand order repaired and checked by reading, not compiled, not run on hardware
//
#include <hip/hip_runtime.h>
#include <hip/hip_bf16.h>

// ---- problem dims (match reference) ----
#define T_DIM   512
#define B_DIM   256
#define OBS_DIM 256
#define D_DIM   512
#define H_DIM   512

#define GRU_BLOCKS 64   // (D/64) * (B/32) persistent blocks, one per WGP

typedef __attribute__((ext_vector_type(16))) __bf16   v16bf;
typedef __attribute__((ext_vector_type(8)))  __bf16   v8bf;
typedef __attribute__((ext_vector_type(8)))  float    v8f;
typedef __attribute__((ext_vector_type(4)))  unsigned u32x4;
typedef __attribute__((ext_vector_type(8)))  unsigned u32x8;

// -------- WMMA helpers (CDNA5 gfx1250, wave32) --------

__device__ inline v16bf cat8(v8bf lo, v8bf hi) {
  return __builtin_shufflevector(lo, hi, 0,1,2,3,4,5,6,7,8,9,10,11,12,13,14,15);
}

__device__ inline v8f wmma_bf16(v16bf a, v16bf b, v8f c) {
  // D = A(16x32 bf16) * B(32x16 bf16) + C(16x16 f32)
  return __builtin_amdgcn_wmma_f32_16x16x32_bf16(
      /*neg_a=*/false, a, /*neg_b=*/false, b,
      /*c_mod=*/(short)0, c, /*reuse_a=*/false, /*reuse_b=*/false);
}

// A-matrix (16x32 bf16, row-major [M,K]) fragment:
// lanes 0-15: row=lane,   K = {k0..k0+7} and {k0+16..k0+23}
// lanes16-31: row=lane-16,K = {k0+8..+15} and {k0+24..+31}
__device__ inline v16bf load_a_bf16(const __bf16* A, int K, int m0, int k0, int lane) {
  int row = lane & 15;
  int kb  = (lane < 16) ? 0 : 8;
  const __bf16* p = A + (size_t)(m0 + row) * K + (k0 + kb);
  v8bf lo = *reinterpret_cast<const v8bf*>(p);
  v8bf hi = *reinterpret_cast<const v8bf*>(p + 16);
  return cat8(lo, hi);
}

// B-matrix (32x16 bf16) fragment from PRE-TRANSPOSED weights Bt[N,K]:
// lanes 0-15: col=lane,   K = {k0..k0+7} and {k0+16..k0+23}
// lanes16-31: col=lane-16,K = {k0+8..+15} and {k0+24..+31}
__device__ inline v16bf load_b_bf16t(const __bf16* Bt, int K, int nrow0, int k0, int lane) {
  int col = lane & 15;
  int kb  = (lane < 16) ? 0 : 8;
  const __bf16* p = Bt + (size_t)(nrow0 + col) * K + (k0 + kb);
  v8bf lo = *reinterpret_cast<const v8bf*>(p);
  v8bf hi = *reinterpret_cast<const v8bf*>(p + 16);
  return cat8(lo, hi);
}

// -------- elementwise convert: f32 -> bf16 --------
__global__ void cvt_bf16(const float* __restrict__ in, __bf16* __restrict__ out, int n) {
  int idx = blockIdx.x * blockDim.x + threadIdx.x;
  if (idx < n) out[idx] = (__bf16)in[idx];
}

// -------- weight convert + transpose: f32 [K,N] -> bf16 [N,K] --------
__global__ void cvt_transpose_bf16(const float* __restrict__ in, __bf16* __restrict__ out,
                                   int K, int N) {
  int idx = blockIdx.x * blockDim.x + threadIdx.x;
  if (idx >= K * N) return;
  int k = idx / N;
  int n = idx - k * N;
  out[(size_t)n * K + k] = (__bf16)in[idx];
}

// -------- init: zero grid barrier + pre-masked bf16 hstate (mask = done[0]) --------
__global__ void gru_init(const float* __restrict__ hstate,
                         const unsigned char* __restrict__ done0,
                         __bf16* __restrict__ h_bf0,
                         unsigned int* __restrict__ barrier_cnt) {
  int idx = blockIdx.x * blockDim.x + threadIdx.x;
  if (idx == 0) *barrier_cnt = 0u;
  if (idx < B_DIM * D_DIM) {
    int b = idx / D_DIM;
    h_bf0[idx] = done0[b] ? (__bf16)0.0f : (__bf16)hstate[idx];
  }
}

// -------- generic WMMA GEMM: Out[M,N] = act(Abf[M,K] * Bt[N,K]^T + bias) --------
// block = 256 threads = 8 waves; block tile 128(M) x 64(N); wave tile 16 x 64.
template<bool RELU, bool OUT_BF16>
__global__ void __launch_bounds__(256) gemm_wmma_bf16(
    const __bf16* __restrict__ A, const __bf16* __restrict__ Bt,
    const float* __restrict__ bias, void* __restrict__ Out,
    int N, int K)
{
  const int lane = threadIdx.x & 31;
  const int wave = threadIdx.x >> 5;
  const int m0 = blockIdx.y * 128 + wave * 16;
  const int n0 = blockIdx.x * 64;

  v8f acc[4] = {};
  for (int k0 = 0; k0 < K; k0 += 32) {
    if (k0 + 32 < K) {  // stream-ahead on the activation operand (global_prefetch_b8)
      __builtin_prefetch(A + (size_t)(m0 + (lane & 15)) * K + k0 + 32, 0, 0);
    }
    v16bf a = load_a_bf16(A, K, m0, k0, lane);
#pragma unroll
    for (int nn = 0; nn < 4; ++nn) {
      v16bf b = load_b_bf16t(Bt, K, n0 + nn * 16, k0, lane);
      acc[nn] = wmma_bf16(a, b, acc[nn]);
    }
  }

  // C/D layout: lanes 0-15 -> N=lane, M=r ; lanes 16-31 -> N=lane-16, M=8+r
  const int n_lane = lane & 15;
  const int m_base = (lane < 16) ? 0 : 8;
#pragma unroll
  for (int nn = 0; nn < 4; ++nn) {
    int n = n0 + nn * 16 + n_lane;
    float bv = bias[n];
#pragma unroll
    for (int r = 0; r < 8; ++r) {
      int m = m0 + m_base + r;
      float v = acc[nn][r] + bv;
      if (RELU) v = v > 0.0f ? v : 0.0f;
      if (OUT_BF16) ((__bf16*)Out)[(size_t)m * N + n] = (__bf16)v;
      else          ((float*) Out)[(size_t)m * N + n] = v;
    }
  }
}

// -------- persistent GRU scan --------
// Exactly GRU_BLOCKS blocks loop over all T steps with a device-wide barrier.
// Each block owns a 32(B) x 64(D) output tile; its 64-column slice of Wh^T
// (3 gates x 64 rows x 512 K = 192KB bf16) is DMA'd into LDS once by the
// Tensor Data Mover and read from LDS for the whole scan. The `done` reset
// mask is pre-applied to the bf16 h ping-pong buffer by the *previous* step,
// so the MAC loop has zero masking work.
__global__ void __launch_bounds__(256) gru_scan_persistent(
    const float* __restrict__ xg,          // [T, B, 3D]
    const unsigned char* __restrict__ done,// [T, B]
    const float* __restrict__ hstate,      // [B, D] f32
    const __bf16* __restrict__ Wht,        // [3D, D] transposed bf16
    const float* __restrict__ bhn,         // [D]
    float* __restrict__ ys,                // [T, B, D] f32 (h history)
    __bf16* __restrict__ ys_bf,            // [T, B, D] bf16 (feeds critic GEMM)
    __bf16* __restrict__ hbuf0,            // [B, D] bf16 ping (pre-masked A operand)
    __bf16* __restrict__ hbuf1,            // [B, D] bf16 pong
    unsigned int* __restrict__ barrier_cnt)
{
  extern __shared__ char smem_raw[];
  __bf16* smem = (__bf16*)smem_raw;      // [3][64][512]

  const int lane = threadIdx.x & 31;
  const int wave = threadIdx.x >> 5;
  const int nblk = blockIdx.x & 7;       // D / 64
  const int mblk = blockIdx.x >> 3;      // B / 32
  const int msub = wave & 1;
  const int nsub = wave >> 1;
  const int m0 = mblk * 32 + msub * 16;
  const int n_base = nblk * 64;

  // ---- TDM: stage this block's Wh^T slice into LDS (one async tensor DMA) ----
  // 3D tile: dim0 = 512 cols (contiguous), dim1 = 64 rows (stride 512),
  //          dim2 = 3 gates (stride 512*512). data_size = 2 bytes.
  if (threadIdx.x < 32) {                // wave 0 issues; TDM ignores EXEC
    unsigned long long ga =
        (unsigned long long)(uintptr_t)(Wht + (size_t)n_base * D_DIM);
    unsigned lds_off = (unsigned)(uintptr_t)smem_raw;   // LDS byte offset
    u32x4 g0;
    g0[0] = 1u;                                          // count=1 valid descriptor
    g0[1] = lds_off;                                     // lds_addr [63:32]
    g0[2] = (unsigned)(ga & 0xffffffffull);              // global_addr [95:64]
    g0[3] = (unsigned)((ga >> 32) & 0x01ffffffull)       // global_addr [120:96]
            | (2u << 30);                                // type=2 ("image") [127:126]
    u32x8 g1;
    g1[0] = 0x00010000u;                 // workgroup_mask=0, data_size=1 (2B)
    g1[1] = (unsigned)D_DIM << 16;       // tensor_dim0 = 512 (bits 63:48)
    g1[2] = (unsigned)D_DIM << 16;       // tensor_dim1 = 512 (bits 95:80)
    g1[3] = (unsigned)D_DIM << 16;       // tile_dim0   = 512 (bits 127:112)
    g1[4] = (3u << 16) | 64u;            // tile_dim1=64, tile_dim2=3
    g1[5] = (unsigned)D_DIM;             // tensor_dim0_stride = 512
    g1[6] = 0u;
    g1[7] = (unsigned)((D_DIM * D_DIM) >> 16);  // tensor_dim1_stride = 512*512
    u32x4 g2;
    g2[0] = 3u;                          // tensor_dim2 = 3
    g2[1] = 0u; g2[2] = 0u; g2[3] = 0u;
    u32x4 g3;
    g3[0] = 0u; g3[1] = 0u; g3[2] = 0u; g3[3] = 0u;
    asm volatile("tensor_load_to_lds %0, %1, %2, %3"
                 :: "s"(g0), "s"(g1), "s"(g2), "s"(g3)
                 : "memory");
    __builtin_amdgcn_s_wait_tensorcnt(0);
  }
  __syncthreads();

  const int arow   = lane & 15;
  const int kb_a   = (lane < 16) ? 0 : 8;
  const int col    = lane & 15;
  const int kb_b   = (lane < 16) ? 0 : 8;
  const int n      = n_base + nsub * 16 + col;   // global output column
  const int m_base = (lane < 16) ? 0 : 8;
  const float bhn_v = bhn[n];
  const int b_a = m0 + arow;                     // batch row of this lane's A data
  const __bf16* smem_b0 = smem + (size_t)(0 * 64 + nsub * 16 + col) * D_DIM + kb_b;
  const __bf16* smem_b1 = smem + (size_t)(1 * 64 + nsub * 16 + col) * D_DIM + kb_b;
  const __bf16* smem_b2 = smem + (size_t)(2 * 64 + nsub * 16 + col) * D_DIM + kb_b;

  for (int t = 0; t < T_DIM; ++t) {
    const __bf16* hA = (t & 1) ? hbuf1 : hbuf0;  // pre-masked bf16 h
    const float*  hF = (t == 0) ? hstate : (ys + (size_t)(t - 1) * B_DIM * D_DIM);
    const unsigned char* dn = done + (size_t)t * B_DIM;
    const float* xg_t = xg + (size_t)t * B_DIM * 3 * D_DIM;

    // ---- MAC loop: A from global (pre-masked bf16), B from LDS ----
    v8f accr = {}, accz = {}, accn = {};
    const __bf16* ap = hA + (size_t)b_a * D_DIM + kb_a;
#pragma unroll 4
    for (int k0 = 0; k0 < D_DIM; k0 += 32) {
      v16bf a = cat8(*reinterpret_cast<const v8bf*>(ap + k0),
                     *reinterpret_cast<const v8bf*>(ap + k0 + 16));
      v16bf br = cat8(*reinterpret_cast<const v8bf*>(smem_b0 + k0),
                      *reinterpret_cast<const v8bf*>(smem_b0 + k0 + 16));
      v16bf bz = cat8(*reinterpret_cast<const v8bf*>(smem_b1 + k0),
                      *reinterpret_cast<const v8bf*>(smem_b1 + k0 + 16));
      v16bf bn = cat8(*reinterpret_cast<const v8bf*>(smem_b2 + k0),
                      *reinterpret_cast<const v8bf*>(smem_b2 + k0 + 16));
      accr = wmma_bf16(a, br, accr);
      accz = wmma_bf16(a, bz, accz);
      accn = wmma_bf16(a, bn, accn);
    }

    // ---- prefetch next step's xg rows while we finish this step ----
    if (t + 1 < T_DIM) {
      const float* xg_n = xg_t + (size_t)B_DIM * 3 * D_DIM;
#pragma unroll
      for (int r = 0; r < 8; ++r) {
        const float* row = xg_n + (size_t)(m0 + m_base + r) * (3 * D_DIM) + n;
        __builtin_prefetch(row, 0, 0);
        __builtin_prefetch(row + D_DIM, 0, 0);
        __builtin_prefetch(row + 2 * D_DIM, 0, 0);
      }
    }

    // ---- gate epilogue (f32) ----
    float*  ho    = ys    + (size_t)t * B_DIM * D_DIM;
    __bf16* hob   = ys_bf + (size_t)t * B_DIM * D_DIM;
    __bf16* hnext = (t & 1) ? hbuf0 : hbuf1;     // consumed by step t+1
    const unsigned char* dn1 = done + (size_t)(t + 1) * B_DIM;  // next step's mask
#pragma unroll
    for (int r = 0; r < 8; ++r) {
      int b = m0 + m_base + r;
      const float* xgrow = xg_t + (size_t)b * (3 * D_DIM);
      float xr = xgrow[n];
      float xz = xgrow[D_DIM + n];
      float xn = xgrow[2 * D_DIM + n];
      float rg = 1.0f / (1.0f + __expf(-(xr + accr[r])));
      float zg = 1.0f / (1.0f + __expf(-(xz + accz[r])));
      float ng = tanhf(xn + rg * (accn[r] + bhn_v));
      float hp = dn[b] ? 0.0f : hF[(size_t)b * D_DIM + n];
      float hnew = (1.0f - zg) * ng + zg * hp;
      size_t idx = (size_t)b * D_DIM + n;
      ho [idx] = hnew;
      hob[idx] = (__bf16)hnew;
      if (t + 1 < T_DIM)                         // pre-mask with done[t+1]
        hnext[idx] = dn1[b] ? (__bf16)0.0f : (__bf16)hnew;
    }

    // ---- device-wide step barrier (release -> arrive -> wait -> acquire) ----
    if (t + 1 < T_DIM) {
      __builtin_amdgcn_fence(__ATOMIC_RELEASE, "agent");   // global_wb
      __syncthreads();
      if (threadIdx.x == 0) {
        __hip_atomic_fetch_add(barrier_cnt, 1u, __ATOMIC_RELAXED, __HIP_MEMORY_SCOPE_AGENT);
        const unsigned target = (unsigned)(t + 1) * (unsigned)GRU_BLOCKS;
        while (__hip_atomic_load(barrier_cnt, __ATOMIC_RELAXED, __HIP_MEMORY_SCOPE_AGENT) < target)
          __builtin_amdgcn_s_sleep(1);
      }
      __syncthreads();
      __builtin_amdgcn_fence(__ATOMIC_ACQUIRE, "agent");   // global_inv
    }
  }
}

// -------- final projection: value[row] = dot(critic[row,:], W2) + b2 --------
__global__ void value_kernel(const __bf16* __restrict__ critic,
                             const float* __restrict__ W2,
                             const float* __restrict__ b2,
                             float* __restrict__ out_value) {
  int row = blockIdx.x * blockDim.x + threadIdx.x;  // 0 .. T*B-1
  const v8bf* c8 = reinterpret_cast<const v8bf*>(critic + (size_t)row * H_DIM);
  float acc = b2[0];
#pragma unroll 4
  for (int i = 0; i < H_DIM / 8; ++i) {
    v8bf c = c8[i];
#pragma unroll
    for (int j = 0; j < 8; ++j) acc += (float)c[j] * W2[i * 8 + j];
  }
  out_value[row] = acc;
}

// ======================= host-side orchestration =======================
extern "C" void kernel_launch(void* const* d_in, const int* in_sizes, int n_in,
                              void* d_out, int out_size, void* d_ws, size_t ws_size,
                              hipStream_t stream) {
  (void)in_sizes; (void)n_in; (void)out_size; (void)ws_size;

  const float*         obs    = (const float*)d_in[0];         // [T,B,OBS]
  const unsigned char* done   = (const unsigned char*)d_in[1]; // [T,B] bool
  const float*         hstate = (const float*)d_in[2];         // [B,D]
  const float*         W_emb  = (const float*)d_in[3];         // [OBS,D]
  const float*         b_emb  = (const float*)d_in[4];         // [D]
  const float*         Wi     = (const float*)d_in[5];         // [D,3D]
  const float*         bi     = (const float*)d_in[6];         // [3D]
  const float*         Wh     = (const float*)d_in[7];         // [D,3D]
  const float*         bhn    = (const float*)d_in[8];         // [D]
  const float*         W1     = (const float*)d_in[9];         // [D,H]
  const float*         b1     = (const float*)d_in[10];        // [H]
  const float*         W2     = (const float*)d_in[11];        // [H,1]
  const float*         b2     = (const float*)d_in[12];        // [1]

  const size_t TB = (size_t)T_DIM * B_DIM;

  char* ws = (char*)d_ws;
  size_t off = 0;
  auto alloc = [&](size_t bytes) -> void* {
    off = (off + 255) & ~(size_t)255;
    void* p = ws + off;
    off += bytes;
    return p;
  };
  unsigned int* bar_cnt = (unsigned int*)alloc(256);
  __bf16* Wemb_t   = (__bf16*)alloc((size_t)OBS_DIM * D_DIM * 2);   // [D,   OBS]
  __bf16* Wi_t     = (__bf16*)alloc((size_t)D_DIM * 3 * D_DIM * 2); // [3D,  D]
  __bf16* Wh_t     = (__bf16*)alloc((size_t)D_DIM * 3 * D_DIM * 2); // [3D,  D]
  __bf16* W1_t     = (__bf16*)alloc((size_t)D_DIM * H_DIM * 2);     // [H,   D]
  __bf16* hbuf0    = (__bf16*)alloc((size_t)B_DIM * D_DIM * 2);     // h ping (pre-masked)
  __bf16* hbuf1    = (__bf16*)alloc((size_t)B_DIM * D_DIM * 2);     // h pong
  __bf16* obs_bf   = (__bf16*)alloc(TB * OBS_DIM * 2);              // [T*B, OBS]
  __bf16* emb_bf   = (__bf16*)alloc(TB * D_DIM * 2);                // [T*B, D]
  float*  xg       = (float*) alloc(TB * 3 * D_DIM * 4);            // [T*B, 3D]
  float*  ys       = (float*) alloc(TB * D_DIM * 4);                // [T*B, D] f32
  __bf16* ys_bf    = (__bf16*)alloc(TB * D_DIM * 2);                // [T*B, D] bf16
  __bf16* critic   = (__bf16*)alloc(TB * H_DIM * 2);                // [T*B, H]

  const dim3 blk(256);

  // --- one-time converts ---
  {
    int n;
    n = OBS_DIM * D_DIM;
    cvt_transpose_bf16<<<(n + 255) / 256, blk, 0, stream>>>(W_emb, Wemb_t, OBS_DIM, D_DIM);
    n = D_DIM * 3 * D_DIM;
    cvt_transpose_bf16<<<(n + 255) / 256, blk, 0, stream>>>(Wi, Wi_t, D_DIM, 3 * D_DIM);
    cvt_transpose_bf16<<<(n + 255) / 256, blk, 0, stream>>>(Wh, Wh_t, D_DIM, 3 * D_DIM);
    n = D_DIM * H_DIM;
    cvt_transpose_bf16<<<(n + 255) / 256, blk, 0, stream>>>(W1, W1_t, D_DIM, H_DIM);
    n = (int)(TB * OBS_DIM);
    cvt_bf16<<<(n + 255) / 256, blk, 0, stream>>>(obs, obs_bf, n);
    gru_init<<<(B_DIM * D_DIM + 255) / 256, blk, 0, stream>>>(hstate, done, hbuf0, bar_cnt);
  }

  // --- emb = relu(obs @ W_emb + b_emb) -> bf16 [T*B, D] ---
  {
    dim3 grid(D_DIM / 64, (unsigned)(TB / 128));
    gemm_wmma_bf16<true, true><<<grid, blk, 0, stream>>>(
        obs_bf, Wemb_t, b_emb, emb_bf, D_DIM, OBS_DIM);
  }

  // --- xg = emb @ Wi + bi -> f32 [T*B, 3D] ---
  {
    dim3 grid((3 * D_DIM) / 64, (unsigned)(TB / 128));
    gemm_wmma_bf16<false, false><<<grid, blk, 0, stream>>>(
        emb_bf, Wi_t, bi, xg, 3 * D_DIM, D_DIM);
  }

  // --- persistent GRU scan: 64 blocks, TDM-staged Wh slice in LDS ---
  {
    const size_t lds_bytes = (size_t)3 * 64 * D_DIM * sizeof(__bf16);  // 192 KB
    gru_scan_persistent<<<GRU_BLOCKS, blk, lds_bytes, stream>>>(
        xg, done, hstate, Wh_t, bhn, ys, ys_bf, hbuf0, hbuf1, bar_cnt);
  }

  // --- critic = relu(ys @ W1 + b1) -> bf16 [T*B, H] ---
  {
    dim3 grid(H_DIM / 64, (unsigned)(TB / 128));
    gemm_wmma_bf16<true, true><<<grid, blk, 0, stream>>>(
        ys_bf, W1_t, b1, critic, H_DIM, D_DIM);
  }

  // --- outputs: d_out = [ h_last (B*D) | value (T*B) ] ---
  float* out_h = (float*)d_out;
  float* out_v = out_h + (size_t)B_DIM * D_DIM;
  value_kernel<<<(unsigned)(TB / 256), blk, 0, stream>>>(critic, W2, b2, out_v);
  hipMemcpyAsync(out_h, ys + (size_t)(T_DIM - 1) * B_DIM * D_DIM,
                 (size_t)B_DIM * D_DIM * sizeof(float),
                 hipMemcpyDeviceToDevice, stream);
}
